// DualInterAttention_59614146068752
// MI455X (gfx1250) — matrix operand order repaired and checked by reading, not compiled, not run on hardware
//
#include <hip/hip_runtime.h>

#define NB   1024
#define LQn  11
#define LKn  34
#define HD   512
#define NH   8
#define DH   64
#define WTH  0.13533528323661270f   // exp(-2)
#define NEGF (-1e30f)

typedef __bf16 bf16;
typedef __attribute__((ext_vector_type(16))) __bf16 v16bf;
typedef __attribute__((ext_vector_type(8)))  __bf16 v8bf;
typedef __attribute__((ext_vector_type(8)))  float  v8f;

// ---------- helpers ----------
__device__ __forceinline__ bf16 f2bf(float f) {
  unsigned u = __float_as_uint(f);
  unsigned r = (u + 0x7FFFu + ((u >> 16) & 1u)) >> 16;
  unsigned short s = (unsigned short)r;
  return __builtin_bit_cast(bf16, s);
}

__device__ __forceinline__ v16bf load_frag16(const bf16* p0, const bf16* p1) {
  v8bf lo = *(const v8bf*)p0;
  v8bf hi = *(const v8bf*)p1;
  return __builtin_shufflevector(lo, hi, 0,1,2,3,4,5,6,7,8,9,10,11,12,13,14,15);
}

// ---------- conversion kernels ----------
__global__ __launch_bounds__(256) void cvt_bf16_kernel(const float* __restrict__ src,
                                                       bf16* __restrict__ dst, long n) {
  long i = (long)blockIdx.x * blockDim.x + threadIdx.x;
  long stride = (long)gridDim.x * blockDim.x;
  for (; i < n; i += stride) dst[i] = f2bf(src[i]);
}

// W [K=512][N=512] f32 -> Wt [N][K] bf16
__global__ __launch_bounds__(256) void wt_cvt_kernel(const float* __restrict__ W,
                                                     bf16* __restrict__ Wt) {
  int idx = blockIdx.x * blockDim.x + threadIdx.x;   // 262144 total
  int n = idx >> 9, k = idx & 511;
  Wt[(size_t)n * 512 + k] = f2bf(W[(size_t)k * 512 + n]);
}

// ---------- WMMA GEMM: C[M,512] = A_bf16[M,512] @ Wt_bf16[512(N),512(K)]^T ----------
__global__ __launch_bounds__(128) void gemm_bf16_kernel(const bf16* __restrict__ A,
                                                        const bf16* __restrict__ Bt,
                                                        float* __restrict__ C, int M) {
  (void)M;
  const int K = 512, N = 512;
  const int lane = threadIdx.x & 31;
  const int wave = threadIdx.x >> 5;
  const int m0 = blockIdx.x * 32;                 // rows m0..m0+31
  const int n0 = blockIdx.y * 256 + wave * 64;    // cols n0..n0+63
  const int lm    = lane & 15;
  const int hiH   = lane >> 4;                    // 0 | 1
  const int khA   = hiH * 16;                     // A: K 0-15 / 16-31 contiguous
  const int khB   = hiH * 16;                     // B: K 0-15 / 16-31 contiguous

  v8f acc[2][4];
  #pragma unroll
  for (int i = 0; i < 2; ++i)
    #pragma unroll
    for (int j = 0; j < 4; ++j)
      #pragma unroll
      for (int r = 0; r < 8; ++r) acc[i][j][r] = 0.0f;

  const bf16* arow0 = A + (size_t)(m0 + lm) * K;
  const bf16* arow1 = A + (size_t)(m0 + 16 + lm) * K;
  const bf16* brow0 = Bt + (size_t)(n0 +  0 + lm) * K;
  const bf16* brow1 = Bt + (size_t)(n0 + 16 + lm) * K;
  const bf16* brow2 = Bt + (size_t)(n0 + 32 + lm) * K;
  const bf16* brow3 = Bt + (size_t)(n0 + 48 + lm) * K;

  for (int kk = 0; kk < K; kk += 32) {
    const int ka = kk + khA;
    const int kb = kk + khB;
    v16bf a0 = load_frag16(arow0 + ka, arow0 + ka + 8);
    v16bf a1 = load_frag16(arow1 + ka, arow1 + ka + 8);
    v16bf b0 = load_frag16(brow0 + kb, brow0 + kb + 8);
    v16bf b1 = load_frag16(brow1 + kb, brow1 + kb + 8);
    v16bf b2 = load_frag16(brow2 + kb, brow2 + kb + 8);
    v16bf b3 = load_frag16(brow3 + kb, brow3 + kb + 8);

    acc[0][0] = __builtin_amdgcn_wmma_f32_16x16x32_bf16(false, a0, false, b0, (short)0, acc[0][0], false, false);
    acc[0][1] = __builtin_amdgcn_wmma_f32_16x16x32_bf16(false, a0, false, b1, (short)0, acc[0][1], false, false);
    acc[0][2] = __builtin_amdgcn_wmma_f32_16x16x32_bf16(false, a0, false, b2, (short)0, acc[0][2], false, false);
    acc[0][3] = __builtin_amdgcn_wmma_f32_16x16x32_bf16(false, a0, false, b3, (short)0, acc[0][3], false, false);
    acc[1][0] = __builtin_amdgcn_wmma_f32_16x16x32_bf16(false, a1, false, b0, (short)0, acc[1][0], false, false);
    acc[1][1] = __builtin_amdgcn_wmma_f32_16x16x32_bf16(false, a1, false, b1, (short)0, acc[1][1], false, false);
    acc[1][2] = __builtin_amdgcn_wmma_f32_16x16x32_bf16(false, a1, false, b2, (short)0, acc[1][2], false, false);
    acc[1][3] = __builtin_amdgcn_wmma_f32_16x16x32_bf16(false, a1, false, b3, (short)0, acc[1][3], false, false);
  }

  // C/D layout: element r of lane L -> row (r + 8*(L>=16)), col = L%16 (within 16x16 tile)
  #pragma unroll
  for (int i = 0; i < 2; ++i) {
    const int mbase = m0 + i * 16 + hiH * 8;
    #pragma unroll
    for (int j = 0; j < 4; ++j) {
      const int n = n0 + j * 16 + lm;
      #pragma unroll
      for (int r = 0; r < 8; ++r)
        C[(size_t)(mbase + r) * N + n] = acc[i][j][r];
    }
  }
}

// ---------- fused per-(b,h) attention ----------
__global__ __launch_bounds__(128) void attn_fused_kernel(
    const float* __restrict__ gQ,  const float* __restrict__ gK,
    const float* __restrict__ gV,  const float* __restrict__ gQv,
    const float* __restrict__ gKv, const float* __restrict__ gVv,
    const unsigned char* __restrict__ pepM, const unsigned char* __restrict__ hlaM,
    const float* __restrict__ aminoM, const float* __restrict__ lenQ,
    bf16* __restrict__ ctxO, bf16* __restrict__ ctxvO,
    float* __restrict__ outA, float* __restrict__ outAv, float* __restrict__ outF5)
{
  __shared__ float sQ[LQn][65], sKv[LQn][65], sVv[LQn][65];
  __shared__ float sK[LKn][65], sV[LKn][65], sQv[LKn][65];
  __shared__ float sAm[LQn][LKn];
  __shared__ float sAmv[LKn][12];
  __shared__ float sSp[LQn][LKn];
  __shared__ float sAttn[LQn][LKn];
  __shared__ float sAttnV[LKn][12];
  __shared__ float sQp[3][LQn];
  __shared__ float sS[LQn], sSv[LKn];
  __shared__ unsigned char sBase[LQn][LKn];
  __shared__ unsigned char sBaseV[LKn][LQn];

  const int tid = threadIdx.x;
  const int bh  = blockIdx.x;
  const int b   = bh >> 3;
  const int h   = bh & 7;

  const size_t qbase = ((size_t)b * LQn) * HD + (size_t)h * DH;
  const size_t kbase = ((size_t)b * LKn) * HD + (size_t)h * DH;
  const size_t mbase = (size_t)b * (LQn * LKn);

  for (int idx = tid; idx < LQn * DH; idx += 128) {
    int q = idx >> 6, d = idx & 63;
    size_t g = qbase + (size_t)q * HD + d;
    sQ[q][d] = gQ[g]; sKv[q][d] = gKv[g]; sVv[q][d] = gVv[g];
  }
  for (int idx = tid; idx < LKn * DH; idx += 128) {
    int k = idx >> 6, d = idx & 63;
    size_t g = kbase + (size_t)k * HD + d;
    sK[k][d] = gK[g]; sV[k][d] = gV[g]; sQv[k][d] = gQv[g];
  }
  for (int idx = tid; idx < LQn * LKn; idx += 128) {
    int q = idx / LKn, k = idx - q * LKn;
    bool am = aminoM[mbase + idx] < 1.0f;
    sBase[q][k]  = ((pepM[mbase + idx] != 0) || am) ? 1 : 0;
    sBaseV[k][q] = ((hlaM[mbase + (size_t)k * LQn + q] != 0) || am) ? 1 : 0;
  }
  __syncthreads();

  // scores
  for (int idx = tid; idx < LQn * LKn; idx += 128) {
    int q = idx / LKn, k = idx - q * LKn;
    float acc = 0.f;
    #pragma unroll
    for (int d = 0; d < DH; ++d) acc += sQ[q][d] * sK[k][d];
    sAm[q][k] = acc * 0.125f;
  }
  for (int idx = tid; idx < LQn * LKn; idx += 128) {
    int k = idx / LQn, q = idx - k * LQn;
    float acc = 0.f;
    #pragma unroll
    for (int d = 0; d < DH; ++d) acc += sQv[k][d] * sKv[q][d];
    sAmv[k][q] = acc * 0.125f;
  }
  // init qpos
  if (tid < LQn) {
    float lq = lenQ[b] - 1.0f;
    float j  = (float)tid;
    float q1 = 13.f * j / lq;         if (q1 > 13.f) q1 = 1e4f;
    float q2 = 9.f * j / lq + 14.f;   if (q2 > 23.f) q2 = 1e4f;
    float q3 = 33.f - 9.f * j / lq;   if (q3 < 24.f) q3 = 1e4f;
    sQp[0][tid] = q1; sQp[1][tid] = q2; sQp[2][tid] = q3;
  }
  __syncthreads();

  for (int it = 0; it < 3; ++it) {
    // space(qpos)
    for (int idx = tid; idx < LQn * LKn; idx += 128) {
      int q = idx / LKn, k = idx - q * LKn;
      float qc, kp;
      if (k < 4)       { qc = sQp[1][q]; kp = 14.f + (float)k; }
      else if (k < 18) { qc = sQp[0][q]; kp = (float)(k - 4); }
      else if (k < 24) { qc = sQp[1][q]; kp = (float)k; }
      else             { qc = sQp[2][q]; kp = (float)k; }
      float d = qc - kp;
      sSp[q][k] = __expf(-0.5f * d * d);
    }
    __syncthreads();
    if (it == 2) break;
    if (tid < LQn) {
      int q = tid;
      float nq0, nq1, nq2;
      { // segment 1: cols 4..17, kpos 0..13
        float x[14]; float mx = -3.4e38f;
        #pragma unroll
        for (int i = 0; i < 14; ++i) {
          int k = 4 + i;
          bool m = (sSp[q][k] < WTH) || sBase[q][k];
          x[i] = m ? NEGF : sAm[q][k];
          mx = fmaxf(mx, x[i]);
        }
        float sum = 0.f;
        #pragma unroll
        for (int i = 0; i < 14; ++i) { x[i] = __expf(x[i] - mx); sum += x[i]; }
        float inv = 1.f / sum, ps = 0.f, wv = 0.f;
        #pragma unroll
        for (int i = 0; i < 14; ++i) {
          int k = 4 + i;
          bool m = (sSp[q][k] < WTH) || sBase[q][k];
          float p = m ? 0.f : x[i] * inv;
          ps += p; wv += p * (float)i;
        }
        nq0 = wv + (1.f - ps) * sQp[0][q];
      }
      { // segment 2: cols {0..3,18..23}, kpos 14..23
        float x[10]; float mx = -3.4e38f;
        #pragma unroll
        for (int i = 0; i < 10; ++i) {
          int k = (i < 4) ? i : (14 + i);
          bool m = (sSp[q][k] < WTH) || sBase[q][k];
          x[i] = m ? NEGF : sAm[q][k];
          mx = fmaxf(mx, x[i]);
        }
        float sum = 0.f;
        #pragma unroll
        for (int i = 0; i < 10; ++i) { x[i] = __expf(x[i] - mx); sum += x[i]; }
        float inv = 1.f / sum, ps = 0.f, wv = 0.f;
        #pragma unroll
        for (int i = 0; i < 10; ++i) {
          int k = (i < 4) ? i : (14 + i);
          bool m = (sSp[q][k] < WTH) || sBase[q][k];
          float p = m ? 0.f : x[i] * inv;
          ps += p; wv += p * (float)(14 + i);
        }
        nq1 = wv + (1.f - ps) * sQp[1][q];
      }
      { // segment 3: cols 24..33, kpos 24..33
        float x[10]; float mx = -3.4e38f;
        #pragma unroll
        for (int i = 0; i < 10; ++i) {
          int k = 24 + i;
          bool m = (sSp[q][k] < WTH) || sBase[q][k];
          x[i] = m ? NEGF : sAm[q][k];
          mx = fmaxf(mx, x[i]);
        }
        float sum = 0.f;
        #pragma unroll
        for (int i = 0; i < 10; ++i) { x[i] = __expf(x[i] - mx); sum += x[i]; }
        float inv = 1.f / sum, ps = 0.f, wv = 0.f;
        #pragma unroll
        for (int i = 0; i < 10; ++i) {
          int k = 24 + i;
          bool m = (sSp[q][k] < WTH) || sBase[q][k];
          float p = m ? 0.f : x[i] * inv;
          ps += p; wv += p * (float)(24 + i);
        }
        nq2 = wv + (1.f - ps) * sQp[2][q];
      }
      sQp[0][q] = nq0; sQp[1][q] = nq1; sQp[2][q] = nq2;
    }
    __syncthreads();
  }

  // final attn rows (full softmax over 34, mask=base)
  if (tid < LQn) {
    int q = tid;
    float x[LKn]; float mx = -3.4e38f;
    #pragma unroll
    for (int k = 0; k < LKn; ++k) {
      x[k] = sBase[q][k] ? NEGF : (sAm[q][k] + sSp[q][k]);
      mx = fmaxf(mx, x[k]);
    }
    float sum = 0.f;
    #pragma unroll
    for (int k = 0; k < LKn; ++k) { x[k] = __expf(x[k] - mx); sum += x[k]; }
    float inv = 1.f / sum, s = 0.f;
    size_t ob = (size_t)bh * (LQn * LKn) + (size_t)q * LKn;
    #pragma unroll
    for (int k = 0; k < LKn; ++k) {
      float p = sBase[q][k] ? 0.f : x[k] * inv;
      sAttn[q][k] = p; s += p;
      outA[ob + k] = p;
    }
    sS[q] = s;
  }
  // attn_v rows (full softmax over 11, mask=base_v), run on a different wave range
  if (tid >= 64 && tid < 64 + LKn) {
    int k = tid - 64;
    float x[LQn]; float mx = -3.4e38f;
    #pragma unroll
    for (int q = 0; q < LQn; ++q) {
      x[q] = sBaseV[k][q] ? NEGF : (sAmv[k][q] + sSp[q][k]);
      mx = fmaxf(mx, x[q]);
    }
    float sum = 0.f;
    #pragma unroll
    for (int q = 0; q < LQn; ++q) { x[q] = __expf(x[q] - mx); sum += x[q]; }
    float inv = 1.f / sum, s = 0.f;
    size_t ob = (size_t)bh * (LKn * LQn) + (size_t)k * LQn;
    #pragma unroll
    for (int q = 0; q < LQn; ++q) {
      float p = sBaseV[k][q] ? 0.f : x[q] * inv;
      sAttnV[k][q] = p; s += p;
      outAv[ob + q] = p;
    }
    sSv[k] = s;
  }
  __syncthreads();

  // full5 = where(amask, -inf, amino + space)
  for (int idx = tid; idx < LQn * LKn; idx += 128) {
    int q = idx / LKn, k = idx - q * LKn;
    bool am = aminoM[mbase + idx] < 1.0f;
    outF5[(size_t)bh * (LQn * LKn) + idx] = am ? -__builtin_inff() : (sAm[q][k] + sSp[q][k]);
  }
  // ctx = (attn @ V + (1-s)*Q) * s  -> bf16
  for (int idx = tid; idx < LQn * DH; idx += 128) {
    int q = idx >> 6, d = idx & 63;
    float c = 0.f;
    #pragma unroll
    for (int k = 0; k < LKn; ++k) c += sAttn[q][k] * sV[k][d];
    float s = sS[q];
    c = (c + (1.f - s) * sQ[q][d]) * s;
    ctxO[((size_t)(b * LQn + q)) * HD + (size_t)h * DH + d] = f2bf(c);
  }
  // ctx_v = (attn_v @ Vv + (1-sv)*Qv) * sv  -> bf16
  for (int idx = tid; idx < LKn * DH; idx += 128) {
    int k = idx >> 6, d = idx & 63;
    float c = 0.f;
    #pragma unroll
    for (int q = 0; q < LQn; ++q) c += sAttnV[k][q] * sVv[q][d];
    float s = sSv[k];
    c = (c + (1.f - s) * sQv[k][d]) * s;
    ctxvO[((size_t)(b * LKn + k)) * HD + (size_t)h * DH + d] = f2bf(c);
  }
}

// ---------- residual add + LayerNorm over 512 ----------
__global__ __launch_bounds__(256) void add_ln_kernel(const float* __restrict__ t,
                                                     const float* __restrict__ r,
                                                     float* __restrict__ o) {
  const int row = blockIdx.x;
  const int tid = threadIdx.x;
  const size_t base = (size_t)row * 512;
  float x0 = t[base + tid] + r[base + tid];
  float x1 = t[base + tid + 256] + r[base + tid + 256];
  __shared__ float rs[256], rq[256];
  rs[tid] = x0 + x1;
  rq[tid] = x0 * x0 + x1 * x1;
  __syncthreads();
  for (int s = 128; s > 0; s >>= 1) {
    if (tid < s) { rs[tid] += rs[tid + s]; rq[tid] += rq[tid + s]; }
    __syncthreads();
  }
  float mu  = rs[0] * (1.f / 512.f);
  float var = rq[0] * (1.f / 512.f) - mu * mu;
  float inv = rsqrtf(var + 1e-5f);
  o[base + tid]       = (x0 - mu) * inv;
  o[base + tid + 256] = (x1 - mu) * inv;
}

// ---------- launch ----------
extern "C" void kernel_launch(void* const* d_in, const int* in_sizes, int n_in,
                              void* d_out, int out_size, void* d_ws, size_t ws_size,
                              hipStream_t stream) {
  (void)in_sizes; (void)n_in; (void)out_size; (void)ws_size;

  const float* query = (const float*)d_in[0];
  const float* key   = (const float*)d_in[1];
  const unsigned char* pepM = (const unsigned char*)d_in[2];
  const unsigned char* hlaM = (const unsigned char*)d_in[3];
  const float* aminoM = (const float*)d_in[4];
  const float* lenq   = (const float*)d_in[5];
  const float* Wmat[8] = {
    (const float*)d_in[7],  (const float*)d_in[8],  (const float*)d_in[9],
    (const float*)d_in[10], (const float*)d_in[11], (const float*)d_in[12],
    (const float*)d_in[13], (const float*)d_in[14]
  };

  const size_t NQ = (size_t)NB * LQn * HD;   // 5,767,168
  const size_t NK = (size_t)NB * LKn * HD;   // 17,825,792

  char* w = (char*)d_ws;
  size_t o = 0;
  auto take = [&](size_t bytes) -> char* {
    char* p = w + o;
    o = (o + bytes + 255) & ~(size_t)255;
    return p;
  };
  bf16* q_bf = (bf16*)take(NQ * 2);
  bf16* k_bf = (bf16*)take(NK * 2);
  bf16* wt[8];
  for (int i = 0; i < 8; ++i) wt[i] = (bf16*)take((size_t)512 * 512 * 2);
  float* pQ  = (float*)take(NQ * 4);
  float* pK  = (float*)take(NK * 4);
  float* pV  = (float*)take(NK * 4);
  float* pQv = (float*)take(NK * 4);
  float* pKv = (float*)take(NQ * 4);
  float* pVv = (float*)take(NQ * 4);
  // aliases (lifetimes do not overlap in stream order)
  bf16* ctx_bf  = q_bf;   // dead after projection GEMMs
  bf16* ctxv_bf = k_bf;
  float* tmp_q  = pQ;     // dead after attn kernel
  float* tmp_k  = pK;

  float* out_q = (float*)d_out;
  float* out_k = out_q + NQ;
  float* outA  = out_k + NK;
  float* outAv = outA + (size_t)NB * NH * LQn * LKn;
  float* outF5 = outAv + (size_t)NB * NH * LQn * LKn;

  // 1) convert inputs / weights to bf16 (weights transposed to [N][K])
  cvt_bf16_kernel<<<dim3(2048), dim3(256), 0, stream>>>(query, q_bf, (long)NQ);
  cvt_bf16_kernel<<<dim3(4096), dim3(256), 0, stream>>>(key,   k_bf, (long)NK);
  for (int i = 0; i < 8; ++i)
    wt_cvt_kernel<<<dim3(1024), dim3(256), 0, stream>>>(Wmat[i], wt[i]);

  // 2) projection GEMMs (WMMA bf16 -> f32)
  gemm_bf16_kernel<<<dim3(11264/32, 2), dim3(128), 0, stream>>>(q_bf, wt[0], pQ,  11264);
  gemm_bf16_kernel<<<dim3(34816/32, 2), dim3(128), 0, stream>>>(k_bf, wt[1], pK,  34816);
  gemm_bf16_kernel<<<dim3(34816/32, 2), dim3(128), 0, stream>>>(k_bf, wt[2], pV,  34816);
  gemm_bf16_kernel<<<dim3(34816/32, 2), dim3(128), 0, stream>>>(k_bf, wt[3], pQv, 34816);
  gemm_bf16_kernel<<<dim3(11264/32, 2), dim3(128), 0, stream>>>(q_bf, wt[4], pKv, 11264);
  gemm_bf16_kernel<<<dim3(11264/32, 2), dim3(128), 0, stream>>>(q_bf, wt[5], pVv, 11264);

  // 3) fused attention per (b,h)
  attn_fused_kernel<<<dim3(NB * NH), dim3(128), 0, stream>>>(
      pQ, pK, pV, pQv, pKv, pVv, pepM, hlaM, aminoM, lenq,
      ctx_bf, ctxv_bf, outA, outAv, outF5);

  // 4) output projections
  gemm_bf16_kernel<<<dim3(11264/32, 2), dim3(128), 0, stream>>>(ctx_bf,  wt[6], tmp_q, 11264);
  gemm_bf16_kernel<<<dim3(34816/32, 2), dim3(128), 0, stream>>>(ctxv_bf, wt[7], tmp_k, 34816);

  // 5) residual + LayerNorm
  add_ln_kernel<<<dim3(11264), dim3(256), 0, stream>>>(tmp_q, query, out_q);
  add_ln_kernel<<<dim3(34816), dim3(256), 0, stream>>>(tmp_k, key,   out_k);
}
